// CLIPAttention_54571854463669
// MI455X (gfx1250) — matrix operand order repaired and checked by reading, not compiled, not run on hardware
//
#include <hip/hip_runtime.h>

typedef __attribute__((ext_vector_type(16))) _Float16 v16h;
typedef __attribute__((ext_vector_type(8)))  _Float16 v8h;
typedef __attribute__((ext_vector_type(8)))  float    v8f;
typedef __attribute__((ext_vector_type(4)))  int      v4i;

static constexpr int Bb = 8, Tt = 1024, Dd = 1024, Hh = 16, HDd = 64;
static constexpr int Mrows = Bb * Tt;               // 8192
static constexpr float ATT_SCALE = 0.125f;          // HD^-0.5

#if __has_builtin(__builtin_amdgcn_global_load_async_to_lds_b128)
#define HAVE_ASYNC_LDS 1
#endif
#if __has_builtin(__builtin_amdgcn_s_wait_asynccnt)
#define HAVE_WAIT_ASYNC 1
#endif

// 16-byte global -> LDS copy: async on CDNA5 toolchains that expose it,
// VGPR round-trip otherwise. Builtin signature (probe-derived): the pointer
// args are v4i* in AS1 (global) and AS3 (LDS).
static __device__ inline void copy16_g2l(const _Float16* __restrict__ g,
                                         _Float16* __restrict__ l) {
#ifdef HAVE_ASYNC_LDS
  __builtin_amdgcn_global_load_async_to_lds_b128(
      (__attribute__((address_space(1))) v4i*)g,
      (__attribute__((address_space(3))) v4i*)l,
      /*offset=*/0, /*cpol=*/0);
#else
  *(v8h*)l = *(const v8h*)g;
#endif
}
static __device__ inline void wait_async_le4() {
#ifdef HAVE_WAIT_ASYNC
  __builtin_amdgcn_s_wait_asynccnt(4);
#endif
}
static __device__ inline void wait_async_0() {
#ifdef HAVE_WAIT_ASYNC
  __builtin_amdgcn_s_wait_asynccnt(0);
#endif
}

static __device__ inline v16h make16(v8h lo, v8h hi) {
  union { v16h v; v8h h[2]; } u;
  u.h[0] = lo; u.h[1] = hi;
  return u.v;
}

static __device__ inline v8f wmma_f16(v16h a, v16h b, v8f c) {
  return __builtin_amdgcn_wmma_f32_16x16x32_f16(
      false, a, false, b, (short)0, c, false, false);
}

// ---------------------------------------------------------------------------
// Prep kernel A: X fp32 -> f16, elementwise. 8 elems/thread.
// ---------------------------------------------------------------------------
__global__ __launch_bounds__(256) void cvt_x_kernel(
    const float* __restrict__ X, _Float16* __restrict__ Xh) {
  const size_t i = ((size_t)blockIdx.x * 256 + threadIdx.x) * 8;
  union { v8h v; _Float16 e[8]; } t;
#pragma unroll
  for (int e = 0; e < 8; ++e) t.e[e] = (_Float16)X[i + e];
  *(v8h*)&Xh[i] = t.v;
}

// ---------------------------------------------------------------------------
// Prep kernel B: W fp32 [K][N] -> f16 transposed [N][K], 32x32 LDS tiles.
// grid = (32, 32, 4) ; z selects Wq/Wk/Wv/Wo.
// ---------------------------------------------------------------------------
__global__ __launch_bounds__(256) void wt_kernel(
    const float* __restrict__ Wq, const float* __restrict__ Wk,
    const float* __restrict__ Wv, const float* __restrict__ Wo,
    _Float16* __restrict__ Wt) {
  const float* W = (blockIdx.z == 0) ? Wq
                 : (blockIdx.z == 1) ? Wk
                 : (blockIdx.z == 2) ? Wv : Wo;
  _Float16* Ot = Wt + (size_t)blockIdx.z * Dd * Dd;
  __shared__ _Float16 tile[32][33];          // padded: no bank conflicts
  const int c  = threadIdx.x & 31;
  const int r0 = threadIdx.x >> 5;           // 0..7
  const int k0 = blockIdx.y * 32, n0 = blockIdx.x * 32;
#pragma unroll
  for (int i = 0; i < 4; ++i)
    tile[r0 + i * 8][c] = (_Float16)W[(size_t)(k0 + r0 + i * 8) * Dd + n0 + c];
  __syncthreads();
#pragma unroll
  for (int i = 0; i < 4; ++i)
    Ot[(size_t)(n0 + r0 + i * 8) * Dd + k0 + c] = tile[c][r0 + i * 8];
}

// ---------------------------------------------------------------------------
// Tiled f16 GEMM: out[M,N] = Ah[M,K] @ Bt[N,K]^T, epilogue (+bias)*scale.
// Block 256 = 8 waves; tile 128x128, k-step 32, double-buffered LDS with
// async global->LDS staging. Wave grid 2x4, 64x32 per wave = 8 WMMA/k-step.
// ---------------------------------------------------------------------------
template <typename OutT>
static __device__ void gemm_tile_f16(const _Float16* __restrict__ Ah,
                                     const _Float16* __restrict__ Bt,
                                     const float* __restrict__ bias,
                                     float scale, OutT* __restrict__ out) {
  constexpr int N = Dd, K = Dd;
  __shared__ _Float16 As[2][128][32];   // [buf][m][k]
  __shared__ _Float16 Bs[2][128][32];   // [buf][n][k]

  const int tid  = threadIdx.x;
  const int lane = tid & 31;
  const int wid  = tid >> 5;
  const int wm   = wid >> 2;            // 0..1
  const int wn   = wid & 3;             // 0..3
  const int m0   = blockIdx.y * 128;
  const int n0   = blockIdx.x * 128;

  const int srow = tid >> 1;            // 0..127 (A row / B column)
  const int sseg = (tid & 1) * 16;      // 16-half segment

  const _Float16* gA = Ah + (size_t)(m0 + srow) * K + sseg;
  const _Float16* gB = Bt + (size_t)(n0 + srow) * K + sseg;

  // 4 copy ops per thread per stage (2 for A, 2 for B)
  auto stage = [&](int buf, int kk) {
    copy16_g2l(gA + kk,     &As[buf][srow][sseg]);
    copy16_g2l(gA + kk + 8, &As[buf][srow][sseg + 8]);
    copy16_g2l(gB + kk,     &Bs[buf][srow][sseg]);
    copy16_g2l(gB + kk + 8, &Bs[buf][srow][sseg + 8]);
  };

  v8f acc[4][2] = {};
  stage(0, 0);

  int it = 0;
  for (int k0 = 0; k0 < K; k0 += 32, ++it) {
    const int cur = it & 1;
    if (k0 + 32 < K) {
      stage(cur ^ 1, k0 + 32);
      wait_async_le4();                 // prior stage (ours) is complete
    } else {
      wait_async_0();
    }
    __syncthreads();                    // all waves' cur-buffer data visible

    const int kh = (lane >> 4) * 8;     // A-frag K-half per ISA layout
    const int kb = (lane >> 4) * 8;     // B-frag K-half per ISA layout
    v16h a[4], b[2];
#pragma unroll
    for (int i = 0; i < 4; ++i) {
      const _Float16* p = &As[cur][wm * 64 + i * 16 + (lane & 15)][0];
      a[i] = make16(*(const v8h*)(p + kh), *(const v8h*)(p + 16 + kh));
    }
#pragma unroll
    for (int j = 0; j < 2; ++j) {
      const _Float16* p = &Bs[cur][wn * 32 + j * 16 + (lane & 15)][0];
      b[j] = make16(*(const v8h*)(p + kb), *(const v8h*)(p + 16 + kb));
    }
#pragma unroll
    for (int i = 0; i < 4; ++i)
#pragma unroll
      for (int j = 0; j < 2; ++j)
        acc[i][j] = wmma_f16(a[i], b[j], acc[i][j]);
    __syncthreads();                    // compute done before buffer reuse
  }

  const int rb = (lane >> 4) * 8;
#pragma unroll
  for (int i = 0; i < 4; ++i) {
#pragma unroll
    for (int j = 0; j < 2; ++j) {
      const int col = n0 + wn * 32 + j * 16 + (lane & 15);
      const float bv = bias[col];
#pragma unroll
      for (int r = 0; r < 8; ++r) {
        const int row = m0 + wm * 64 + i * 16 + rb + r;
        out[(size_t)row * N + col] = (OutT)((acc[i][j][r] + bv) * scale);
      }
    }
  }
}

// ---------------------------------------------------------------------------
// Kernel: fused QKV projections. grid = (N/128, M/128, 3)
// ---------------------------------------------------------------------------
__global__ __launch_bounds__(256) void qkv_gemm_kernel(
    const _Float16* __restrict__ Xh, const _Float16* __restrict__ Wt,
    const float* __restrict__ bq, const float* __restrict__ bk,
    const float* __restrict__ bv,
    _Float16* __restrict__ Q, _Float16* __restrict__ Kf,
    _Float16* __restrict__ Vf) {
  const size_t wsz = (size_t)Dd * Dd;
  const _Float16* W; const float* bias; _Float16* out; float scale = 1.0f;
  if (blockIdx.z == 0)      { W = Wt;           bias = bq; out = Q;  scale = ATT_SCALE; }
  else if (blockIdx.z == 1) { W = Wt + wsz;     bias = bk; out = Kf; }
  else                      { W = Wt + 2 * wsz; bias = bv; out = Vf; }
  gemm_tile_f16<_Float16>(Xh, W, bias, scale, out);
}

// ---------------------------------------------------------------------------
// Kernel: flash attention. grid = (T/128, B*H); 8 waves, 16 q-rows each.
// ---------------------------------------------------------------------------
__global__ __launch_bounds__(256) void flash_attn_kernel(
    const _Float16* __restrict__ Q, const _Float16* __restrict__ Kf,
    const _Float16* __restrict__ Vf, _Float16* __restrict__ O) {
  __shared__ _Float16 Ks[64][64];       // [key][d]
  __shared__ _Float16 Vt[64][64];       // [d][key]
  __shared__ _Float16 Ps[8][16][64];    // per-wave P transpose buffer

  const int tid  = threadIdx.x;
  const int lane = tid & 31;
  const int wid  = tid >> 5;
  const int bh   = blockIdx.y;
  const int bat  = bh >> 4;
  const int h    = bh & 15;
  const int q0   = blockIdx.x * 128 + wid * 16;
  const size_t rowbase = (size_t)bat * Tt;

  // Q fragments resident in VGPRs for the whole kernel (A layout, d = 0..63)
  const _Float16* qp = Q + (rowbase + q0 + (lane & 15)) * Dd + h * HDd;
  const int kh = (lane >> 4) * 8;
  const v16h qa0 = make16(*(const v8h*)(qp + kh),      *(const v8h*)(qp + 16 + kh));
  const v16h qa1 = make16(*(const v8h*)(qp + 32 + kh), *(const v8h*)(qp + 48 + kh));

  float mrow[8], lrow[8];
#pragma unroll
  for (int r = 0; r < 8; ++r) { mrow[r] = -1e30f; lrow[r] = 0.0f; }
  v8f oacc[4] = {};

  const int skey = tid >> 2;            // key row staged by this thread
  const int sdg  = (tid & 3) * 16;      // 16-wide d group

  for (int kt = 0; kt < Tt; kt += 64) {
    // ---- stage K tile [key][d] (async path) ----
    {
      const _Float16* src = Kf + (rowbase + kt + skey) * Dd + h * HDd + sdg;
      copy16_g2l(src,     &Ks[skey][sdg]);
      copy16_g2l(src + 8, &Ks[skey][sdg + 8]);
    }
    // ---- stage V tile transposed to [d][key] ----
    {
      const _Float16* src = Vf + (rowbase + kt + skey) * Dd + h * HDd + sdg;
      union { v8h v[2]; _Float16 e[16]; } t;
      t.v[0] = *(const v8h*)(src);
      t.v[1] = *(const v8h*)(src + 8);
#pragma unroll
      for (int e = 0; e < 16; ++e) Vt[sdg + e][skey] = t.e[e];
    }
    if (kt + 64 < Tt) {
      __builtin_prefetch(Kf + (rowbase + kt + 64 + skey) * Dd + h * HDd + sdg, 0, 1);
      __builtin_prefetch(Vf + (rowbase + kt + 64 + skey) * Dd + h * HDd + sdg, 0, 1);
    }
    wait_async_0();
    __syncthreads();

    // ---- S = Q @ K^T : 16 x 64 per wave ----
    v8f s[4] = {};
#pragma unroll
    for (int j = 0; j < 4; ++j) {
      const _Float16* kp = &Ks[j * 16 + (lane & 15)][0];
      const int doff = (lane >> 4) * 8;
      const v16h b0 = make16(*(const v8h*)(kp + doff),
                             *(const v8h*)(kp + 16 + doff));
      const v16h b1 = make16(*(const v8h*)(kp + 32 + doff),
                             *(const v8h*)(kp + 48 + doff));
      s[j] = wmma_f16(qa0, b0, s[j]);
      s[j] = wmma_f16(qa1, b1, s[j]);
    }

    // ---- online softmax ----
    const int rb = (lane >> 4) * 8;
#pragma unroll
    for (int r = 0; r < 8; ++r) {
      float vmax = fmaxf(fmaxf(s[0][r], s[1][r]), fmaxf(s[2][r], s[3][r]));
#pragma unroll
      for (int off = 8; off >= 1; off >>= 1)
        vmax = fmaxf(vmax, __shfl_xor(vmax, off, 32));
      const float mnew = fmaxf(mrow[r], vmax);
      const float corr = __expf(mrow[r] - mnew);
      float psum = 0.0f;
#pragma unroll
      for (int j = 0; j < 4; ++j) {
        const float p = __expf(s[j][r] - mnew);
        psum += p;
        Ps[wid][rb + r][j * 16 + (lane & 15)] = (_Float16)p;
      }
#pragma unroll
      for (int off = 8; off >= 1; off >>= 1)
        psum += __shfl_xor(psum, off, 32);
      lrow[r] = lrow[r] * corr + psum;
      mrow[r] = mnew;
#pragma unroll
      for (int dt = 0; dt < 4; ++dt) oacc[dt][r] *= corr;
    }

    // ---- O += P @ V ----
    {
      const _Float16* pr = &Ps[wid][lane & 15][0];
      const int kh8 = (lane >> 4) * 8;
      const v16h pa0 = make16(*(const v8h*)(pr + kh8),
                              *(const v8h*)(pr + 16 + kh8));
      const v16h pa1 = make16(*(const v8h*)(pr + 32 + kh8),
                              *(const v8h*)(pr + 48 + kh8));
#pragma unroll
      for (int dt = 0; dt < 4; ++dt) {
        const _Float16* vp = &Vt[dt * 16 + (lane & 15)][0];
        const int ko = (lane >> 4) * 8;
        const v16h vb0 = make16(*(const v8h*)(vp + ko),
                                *(const v8h*)(vp + 16 + ko));
        const v16h vb1 = make16(*(const v8h*)(vp + 32 + ko),
                                *(const v8h*)(vp + 48 + ko));
        oacc[dt] = wmma_f16(pa0, vb0, oacc[dt]);
        oacc[dt] = wmma_f16(pa1, vb1, oacc[dt]);
      }
    }
    __syncthreads();
  }

  const int rb = (lane >> 4) * 8;
#pragma unroll
  for (int dt = 0; dt < 4; ++dt) {
    const int col = h * HDd + dt * 16 + (lane & 15);
#pragma unroll
    for (int r = 0; r < 8; ++r) {
      const int row = q0 + rb + r;
      O[(rowbase + row) * Dd + col] = (_Float16)(oacc[dt][r] / lrow[r]);
    }
  }
}

// ---------------------------------------------------------------------------
// Kernel: output projection -> fp32 d_out. grid = (N/128, M/128)
// ---------------------------------------------------------------------------
__global__ __launch_bounds__(256) void out_gemm_kernel(
    const _Float16* __restrict__ A, const _Float16* __restrict__ WtO,
    const float* __restrict__ bo, float* __restrict__ out) {
  gemm_tile_f16<float>(A, WtO, bo, 1.0f, out);
}

// ---------------------------------------------------------------------------
extern "C" void kernel_launch(void* const* d_in, const int* in_sizes, int n_in,
                              void* d_out, int out_size, void* d_ws,
                              size_t ws_size, hipStream_t stream) {
  const float* X  = (const float*)d_in[0];
  const float* Wq = (const float*)d_in[1];
  const float* bq = (const float*)d_in[2];
  const float* Wk = (const float*)d_in[3];
  const float* bk = (const float*)d_in[4];
  const float* Wv = (const float*)d_in[5];
  const float* bv = (const float*)d_in[6];
  const float* Wo = (const float*)d_in[7];
  const float* bo = (const float*)d_in[8];
  float* out = (float*)d_out;

  const size_t mat = (size_t)Mrows * Dd;    // 8M elements
  const size_t wsz = (size_t)Dd * Dd;       // 1M elements
  _Float16* Qf = (_Float16*)d_ws;           // 16 MB
  _Float16* Kf = Qf + mat;                  // 16 MB
  _Float16* Vf = Kf + mat;                  // 16 MB
  _Float16* Af = Vf + mat;                  // 16 MB
  _Float16* Xh = Af + mat;                  // 16 MB
  _Float16* Wt = Xh + mat;                  // 4 x 2 MB (q,k,v,o transposed)

  cvt_x_kernel<<<dim3((unsigned)(mat / (256 * 8))), 256, 0, stream>>>(X, Xh);
  wt_kernel<<<dim3(Dd / 32, Dd / 32, 4), 256, 0, stream>>>(Wq, Wk, Wv, Wo, Wt);
  qkv_gemm_kernel<<<dim3(Dd / 128, Mrows / 128, 3), 256, 0, stream>>>(
      Xh, Wt, bq, bk, bv, Qf, Kf, Vf);
  flash_attn_kernel<<<dim3(Tt / 128, Bb * Hh), 256, 0, stream>>>(
      Qf, Kf, Vf, Af);
  out_gemm_kernel<<<dim3(Dd / 128, Mrows / 128), 256, 0, stream>>>(
      Af, Wt + 3 * wsz, bo, out);
}
